// DeepseekV2Attention_48198122995780
// MI455X (gfx1250) — matrix operand order repaired and checked by reading, not compiled, not run on hardware
//
#include <hip/hip_runtime.h>
#include <hip/hip_bf16.h>

typedef __bf16 bf16_t;
typedef __attribute__((ext_vector_type(16))) __bf16 v16bf;
typedef __attribute__((ext_vector_type(8)))  __bf16 v8bf;
typedef __attribute__((ext_vector_type(8)))  float  v8f;

#define H_    16
#define NOPE_ 128
#define ROPE_ 64
#define VD_   128
#define HEAD_ 192
#define SCALE_ 0.07216878364870323f  // 192^-0.5

__device__ __forceinline__ bf16_t f2bf(float f) {
  unsigned u = __float_as_uint(f);
  u += 0x7FFFu + ((u >> 16) & 1u);            // round-to-nearest-even
  unsigned short h = (unsigned short)(u >> 16);
  return __builtin_bit_cast(bf16_t, h);
}

__device__ __forceinline__ v16bf make_frag(v8bf lo, v8bf hi) {
  v16bf r;
#pragma unroll
  for (int i = 0; i < 8; ++i) { r[i] = lo[i]; r[i + 8] = hi[i]; }
  return r;
}

// ---------------------------------------------------------------- elementwise
__global__ void k_cvt_bf16(const float* __restrict__ src, bf16_t* __restrict__ dst, int n) {
  int i = blockIdx.x * blockDim.x + threadIdx.x;
  int stride = gridDim.x * blockDim.x;
  for (; i < n; i += stride) dst[i] = f2bf(src[i]);
}

// ---------------------------------------------------------------- GEMM (bf16 in, f32 out)
// C[M,N] = A[M,K] @ B[K,N].  M%128==0, N%64==0, K%32==0.
// A tile staged global->LDS with GLOBAL_LOAD_ASYNC_TO_LDS_B128 (ASYNCcnt),
// B tile staged transposed ([n][k]) so B fragments are contiguous LDS b128 loads.
__global__ __launch_bounds__(128)
void k_gemm_bf16(const bf16_t* __restrict__ A, const bf16_t* __restrict__ B,
                 float* __restrict__ C, int M, int N, int K) {
  __shared__ __align__(16) bf16_t As[128][32];  // [m][k]
  __shared__ __align__(16) bf16_t Bs[64][32];   // transposed: [n][k]
  (void)M;
  const int nb = N >> 6;
  const int m0 = (blockIdx.x / nb) * 128;
  const int n0 = (blockIdx.x % nb) * 64;
  const int tid  = threadIdx.x;
  const int lane = tid & 31;
  const int wave = tid >> 5;
  const int wm = wave * 32;                 // each wave: 32 rows x 64 cols
  const int lm = lane & 15, hi = lane >> 4;

  v8f acc[2][4];
#pragma unroll
  for (int i = 0; i < 2; ++i)
#pragma unroll
    for (int j = 0; j < 4; ++j)
#pragma unroll
      for (int e = 0; e < 8; ++e) acc[i][j][e] = 0.f;

  for (int k0 = 0; k0 < K; k0 += 32) {
    // A tile: 128x32 bf16 = 8KB, async copy, 4 x 16B per lane
#pragma unroll
    for (int c = 0; c < 4; ++c) {
      int chunk = c * 128 + tid;            // 0..511 16-byte chunks
      int row  = chunk >> 2;
      int col8 = (chunk & 3) * 8;
      unsigned lds = (unsigned)(unsigned long long)&As[row][col8];
      unsigned long long ga =
          (unsigned long long)(A + (size_t)(m0 + row) * K + k0 + col8);
      asm volatile("global_load_async_to_lds_b128 %0, %1, off"
                   :: "v"(lds), "v"(ga) : "memory");
    }
    // B tile: 32x64, stored transposed into Bs[n][k]
    {
      int kr = tid >> 2, cb = (tid & 3) * 16;
      const v8bf* g = (const v8bf*)(B + (size_t)(k0 + kr) * N + n0 + cb);
      v8bf b0 = g[0], b1 = g[1];
#pragma unroll
      for (int i = 0; i < 8; ++i) {
        Bs[cb + i][kr]     = b0[i];
        Bs[cb + 8 + i][kr] = b1[i];
      }
    }
    asm volatile("s_wait_asynccnt 0x0" ::: "memory");
    __syncthreads();

    v16bf af[2], bfr[4];
#pragma unroll
    for (int t = 0; t < 2; ++t) {
      int m = wm + t * 16 + lm;
      af[t] = make_frag(*(const v8bf*)&As[m][hi * 8],
                        *(const v8bf*)&As[m][16 + hi * 8]);
    }
#pragma unroll
    for (int t = 0; t < 4; ++t) {
      int n = t * 16 + lm;
      bfr[t] = make_frag(*(const v8bf*)&Bs[n][hi * 8],
                         *(const v8bf*)&Bs[n][16 + hi * 8]);
    }
#pragma unroll
    for (int i = 0; i < 2; ++i)
#pragma unroll
      for (int j = 0; j < 4; ++j)
        acc[i][j] = __builtin_amdgcn_wmma_f32_16x16x32_bf16(
            false, af[i], false, bfr[j], (short)0, acc[i][j], false, false);
    __syncthreads();
  }
#pragma unroll
  for (int i = 0; i < 2; ++i)
#pragma unroll
    for (int j = 0; j < 4; ++j)
#pragma unroll
      for (int v = 0; v < 8; ++v) {
        int row = m0 + wm + i * 16 + v + hi * 8;
        int col = n0 + j * 16 + lm;
        C[(size_t)row * N + col] = acc[i][j][v];
      }
}

// ---------------------------------------------------------------- RMSNorm (one wave / row)
__global__ __launch_bounds__(32)
void k_rmsnorm_bf16(const float* __restrict__ x, const float* __restrict__ w,
                    bf16_t* __restrict__ out, int ld, int n) {
  int row = blockIdx.x;
  const float* xr = x + (size_t)row * ld;
  int lane = threadIdx.x;
  float s = 0.f;
  for (int j = lane; j < n; j += 32) { float v = xr[j]; s += v * v; }
#pragma unroll
  for (int d = 16; d >= 1; d >>= 1) s += __shfl_xor(s, d, 32);
  float inv = rsqrtf(s / (float)n + 1e-6f);
  bf16_t* orow = out + (size_t)row * n;
  for (int j = lane; j < n; j += 32) orow[j] = f2bf(xr[j] * inv * w[j]);
}

// ---------------------------------------------------------------- RoPE + pack q -> query bf16 (T,H,192), SCALE folded in
__global__ void k_rope_q(const float* __restrict__ q, const float* __restrict__ cosp,
                         const float* __restrict__ sinp, bf16_t* __restrict__ query, int T) {
  int idx = blockIdx.x * blockDim.x + threadIdx.x;
  if (idx >= T * H_ * 32) return;
  int i = idx & 31;
  int h = (idx >> 5) & (H_ - 1);
  int t = idx >> 9;
  const float* qr = q + (size_t)t * (H_ * HEAD_) + h * HEAD_;
  bf16_t* o = query + ((size_t)t * H_ + h) * HEAD_;
  float c = cosp[t * 32 + i], s = sinp[t * 32 + i];
  float e = qr[NOPE_ + 2 * i], od = qr[NOPE_ + 2 * i + 1];
  o[NOPE_ + i]      = f2bf((e * c - od * s) * SCALE_);
  o[NOPE_ + 32 + i] = f2bf((od * c + e * s) * SCALE_);
#pragma unroll
  for (int r = 0; r < 4; ++r) { int d = i * 4 + r; o[d] = f2bf(qr[d] * SCALE_); }
}

// ---------------------------------------------------------------- RoPE k_pe + pack key(T,H,192)/v(T,H,128)
__global__ void k_rope_kv(const float* __restrict__ ckv, const float* __restrict__ kv,
                          const float* __restrict__ cosp, const float* __restrict__ sinp,
                          bf16_t* __restrict__ key, bf16_t* __restrict__ v, int T) {
  int idx = blockIdx.x * blockDim.x + threadIdx.x;
  if (idx >= T * H_ * 32) return;
  int i = idx & 31;
  int h = (idx >> 5) & (H_ - 1);
  int t = idx >> 9;
  float c = cosp[t * 32 + i], s = sinp[t * 32 + i];
  float e  = ckv[(size_t)t * 576 + 512 + 2 * i];
  float od = ckv[(size_t)t * 576 + 512 + 2 * i + 1];
  bf16_t* kr = key + ((size_t)t * H_ + h) * HEAD_;
  bf16_t* vr = v   + ((size_t)t * H_ + h) * VD_;
  const float* kvr = kv + (size_t)t * (H_ * 256) + h * 256;
  kr[NOPE_ + i]      = f2bf(e * c - od * s);
  kr[NOPE_ + 32 + i] = f2bf(od * c + e * s);
#pragma unroll
  for (int r = 0; r < 4; ++r) {
    int d = i * 4 + r;
    kr[d] = f2bf(kvr[d]);
    vr[d] = f2bf(kvr[NOPE_ + d]);
  }
}

// ---------------------------------------------------------------- flash attention, one wave per (qtile16, head)
__global__ __launch_bounds__(32)
void k_flash(const bf16_t* __restrict__ query, const bf16_t* __restrict__ key,
             const bf16_t* __restrict__ v, float* __restrict__ out,
             const int* __restrict__ bptr, int T) {
  __shared__ __align__(16) bf16_t Pl[16][32];    // probs tile (q x key)
  __shared__ __align__(16) bf16_t Vt[128][32];   // V transposed (dim x key)
  const int S = T / *bptr;
  const int hh  = blockIdx.x % H_;
  const int qt  = blockIdx.x / H_;
  const int qr0 = qt * 16;
  const int batch = qr0 / S;
  const int lq0 = qr0 - batch * S;
  const int lane = threadIdx.x;
  const int lm = lane & 15, hi = lane >> 4;

  // Q fragments (6 x 32-dim chunks), resident for whole row block
  v16bf qf[6];
  {
    const bf16_t* qrow = query + ((size_t)(qr0 + lm) * H_ + hh) * HEAD_;
#pragma unroll
    for (int kc = 0; kc < 6; ++kc)
      qf[kc] = make_frag(*(const v8bf*)(qrow + kc * 32 + hi * 8),
                         *(const v8bf*)(qrow + kc * 32 + 16 + hi * 8));
  }

  v8f oacc[8];
#pragma unroll
  for (int nc = 0; nc < 8; ++nc)
#pragma unroll
    for (int e = 0; e < 8; ++e) oacc[nc][e] = 0.f;
  float mrow[8], lrow[8];
#pragma unroll
  for (int i = 0; i < 8; ++i) { mrow[i] = -1e30f; lrow[i] = 0.f; }

  const int kend = (lq0 + 15) / 32;   // inclusive 32-key chunk index
  for (int kc = 0; kc <= kend; ++kc) {
    const int kl0 = kc * 32;
    { // stage V^T for this key chunk: lane <-> key
      int krow = batch * S + kl0 + lane;
      if (krow > T - 1) krow = T - 1;
      const bf16_t* vrow = v + ((size_t)krow * H_ + hh) * VD_;
#pragma unroll
      for (int c8 = 0; c8 < 16; ++c8) {
        v8bf vv = *(const v8bf*)(vrow + c8 * 8);
#pragma unroll
        for (int e = 0; e < 8; ++e) Vt[c8 * 8 + e][lane] = vv[e];
      }
    }
    // scores for two 16-key tiles
    v8f s0, s1;
#pragma unroll
    for (int e = 0; e < 8; ++e) { s0[e] = 0.f; s1[e] = 0.f; }
    {
      int kr0 = batch * S + kl0 + lm;      if (kr0 > T - 1) kr0 = T - 1;
      int kr1 = batch * S + kl0 + 16 + lm; if (kr1 > T - 1) kr1 = T - 1;
      const bf16_t* krow0 = key + ((size_t)kr0 * H_ + hh) * HEAD_;
      const bf16_t* krow1 = key + ((size_t)kr1 * H_ + hh) * HEAD_;
#pragma unroll
      for (int kc6 = 0; kc6 < 6; ++kc6) {
        v16bf b0 = make_frag(*(const v8bf*)(krow0 + kc6 * 32 + hi * 8),
                             *(const v8bf*)(krow0 + kc6 * 32 + 16 + hi * 8));
        v16bf b1 = make_frag(*(const v8bf*)(krow1 + kc6 * 32 + hi * 8),
                             *(const v8bf*)(krow1 + kc6 * 32 + 16 + hi * 8));
        s0 = __builtin_amdgcn_wmma_f32_16x16x32_bf16(false, qf[kc6], false, b0,
                                                     (short)0, s0, false, false);
        s1 = __builtin_amdgcn_wmma_f32_16x16x32_bf16(false, qf[kc6], false, b1,
                                                     (short)0, s1, false, false);
      }
    }
    // causal mask (scale already folded into q)
    if (kl0 + 31 > lq0) {
#pragma unroll
      for (int i = 0; i < 8; ++i) {
        int qrl = lq0 + i + hi * 8;
        if (kl0 + lm > qrl)      s0[i] = -1e30f;
        if (kl0 + 16 + lm > qrl) s1[i] = -1e30f;
      }
    }
    // online softmax (row stats reduced across the 16-lane half)
    float cm[8], al[8], ps[8];
#pragma unroll
    for (int i = 0; i < 8; ++i) cm[i] = fmaxf(s0[i], s1[i]);
#pragma unroll
    for (int d = 1; d < 16; d <<= 1)
#pragma unroll
      for (int i = 0; i < 8; ++i) cm[i] = fmaxf(cm[i], __shfl_xor(cm[i], d, 32));
#pragma unroll
    for (int i = 0; i < 8; ++i) {
      float nm = fmaxf(mrow[i], cm[i]);
      al[i] = __expf(mrow[i] - nm);
      mrow[i] = nm;
      float p0 = __expf(s0[i] - nm);
      float p1 = __expf(s1[i] - nm);
      s0[i] = p0; s1[i] = p1;
      ps[i] = p0 + p1;
    }
#pragma unroll
    for (int d = 1; d < 16; d <<= 1)
#pragma unroll
      for (int i = 0; i < 8; ++i) ps[i] += __shfl_xor(ps[i], d, 32);
#pragma unroll
    for (int i = 0; i < 8; ++i) lrow[i] = lrow[i] * al[i] + ps[i];
#pragma unroll
    for (int nc = 0; nc < 8; ++nc)
#pragma unroll
      for (int i = 0; i < 8; ++i) oacc[nc][i] *= al[i];
    // P: C-layout -> LDS -> A-fragment layout
#pragma unroll
    for (int i = 0; i < 8; ++i) {
      Pl[i + hi * 8][lm]      = f2bf(s0[i]);
      Pl[i + hi * 8][16 + lm] = f2bf(s1[i]);
    }
    __syncthreads();   // single-wave block: cheap LDS fence/ordering
    v16bf pf = make_frag(*(const v8bf*)&Pl[lm][hi * 8],
                         *(const v8bf*)&Pl[lm][16 + hi * 8]);
#pragma unroll
    for (int nc = 0; nc < 8; ++nc) {
      v16bf bv = make_frag(*(const v8bf*)&Vt[nc * 16 + lm][hi * 8],
                           *(const v8bf*)&Vt[nc * 16 + lm][16 + hi * 8]);
      oacc[nc] = __builtin_amdgcn_wmma_f32_16x16x32_bf16(false, pf, false, bv,
                                                         (short)0, oacc[nc], false, false);
    }
    __syncthreads();
  }
  // epilogue: out[T, H*VD] f32
#pragma unroll
  for (int nc = 0; nc < 8; ++nc)
#pragma unroll
    for (int i = 0; i < 8; ++i) {
      int row = qr0 + i + hi * 8;
      int col = hh * VD_ + nc * 16 + lm;
      out[(size_t)row * (H_ * VD_) + col] = oacc[nc][i] / lrow[i];
    }
}

// ---------------------------------------------------------------- launcher
extern "C" void kernel_launch(void* const* d_in, const int* in_sizes, int n_in,
                              void* d_out, int out_size, void* d_ws, size_t ws_size,
                              hipStream_t stream) {
  (void)n_in; (void)out_size; (void)ws_size;
  const float* hidden = (const float*)d_in[0];
  const float* cosp   = (const float*)d_in[1];
  const float* sinp   = (const float*)d_in[2];
  const float* w_q_a  = (const float*)d_in[3];
  const float* q_a_ln = (const float*)d_in[4];
  const float* w_q_b  = (const float*)d_in[5];
  const float* w_kv_a = (const float*)d_in[6];
  const float* kv_ln  = (const float*)d_in[7];
  const float* w_kv_b = (const float*)d_in[8];
  const float* w_o    = (const float*)d_in[9];
  const int*   bptr   = (const int*)d_in[10];

  const int D = 2048;
  const int T = in_sizes[0] / D;         // 4096
  const int QA = 1536, QB = H_ * HEAD_;  // 3072
  const int CKV = 576, KVB = H_ * (NOPE_ + VD_); // 4096
  const int DO = H_ * VD_;               // 2048

  char* ws = (char*)d_ws;
  size_t off = 0;
  auto alloc = [&](size_t bytes) -> void* {
    void* p = (void*)(ws + off);
    off += (bytes + 255) & ~(size_t)255;
    return p;
  };
  bf16_t* hbf     = (bf16_t*)alloc((size_t)T * D * 2);
  bf16_t* wqa_bf  = (bf16_t*)alloc((size_t)D * QA * 2);
  bf16_t* wqb_bf  = (bf16_t*)alloc((size_t)QA * QB * 2);
  bf16_t* wkva_bf = (bf16_t*)alloc((size_t)D * CKV * 2);
  bf16_t* wkvb_bf = (bf16_t*)alloc((size_t)512 * KVB * 2);
  bf16_t* wo_bf   = (bf16_t*)alloc((size_t)DO * D * 2);
  float*  qa_f    = (float*)alloc((size_t)T * QA * 4);
  bf16_t* qan_bf  = (bf16_t*)alloc((size_t)T * QA * 2);
  float*  q_f     = (float*)alloc((size_t)T * QB * 4);
  float*  ckv_f   = (float*)alloc((size_t)T * CKV * 4);
  bf16_t* ckvn_bf = (bf16_t*)alloc((size_t)T * 512 * 2);
  float*  kv_f    = (float*)alloc((size_t)T * KVB * 4);
  bf16_t* query_b = (bf16_t*)alloc((size_t)T * H_ * HEAD_ * 2);
  bf16_t* key_b   = (bf16_t*)alloc((size_t)T * H_ * HEAD_ * 2);
  bf16_t* v_b     = (bf16_t*)alloc((size_t)T * H_ * VD_ * 2);
  float*  attn_f  = (float*)alloc((size_t)T * DO * 4);
  bf16_t* attn_bf = (bf16_t*)alloc((size_t)T * DO * 2);

  auto cvt = [&](const float* s, bf16_t* d, size_t n) {
    int blk = 256;
    int grid = (int)((n + 256 * 8 - 1) / (256 * 8));
    if (grid > 4096) grid = 4096;
    k_cvt_bf16<<<grid, blk, 0, stream>>>(s, d, (int)n);
  };
  auto gemm = [&](const bf16_t* A, const bf16_t* B, float* C, int M, int N, int K) {
    k_gemm_bf16<<<dim3((M / 128) * (N / 64)), 128, 0, stream>>>(A, B, C, M, N, K);
  };

  // 1. bf16 conversions
  cvt(hidden, hbf, (size_t)T * D);
  cvt(w_q_a, wqa_bf, (size_t)D * QA);
  cvt(w_q_b, wqb_bf, (size_t)QA * QB);
  cvt(w_kv_a, wkva_bf, (size_t)D * CKV);
  cvt(w_kv_b, wkvb_bf, (size_t)512 * KVB);
  cvt(w_o, wo_bf, (size_t)DO * D);

  // 2. q path
  gemm(hbf, wqa_bf, qa_f, T, QA, D);
  k_rmsnorm_bf16<<<T, 32, 0, stream>>>(qa_f, q_a_ln, qan_bf, QA, QA);
  gemm(qan_bf, wqb_bf, q_f, T, QB, QA);

  // 3. kv path
  gemm(hbf, wkva_bf, ckv_f, T, CKV, D);
  k_rmsnorm_bf16<<<T, 32, 0, stream>>>(ckv_f, kv_ln, ckvn_bf, CKV, 512);
  gemm(ckvn_bf, wkvb_bf, kv_f, T, KVB, 512);

  // 4. rope + pack
  {
    int n = T * H_ * 32;
    k_rope_q<<<(n + 255) / 256, 256, 0, stream>>>(q_f, cosp, sinp, query_b, T);
    k_rope_kv<<<(n + 255) / 256, 256, 0, stream>>>(ckv_f, kv_f, cosp, sinp, key_b, v_b, T);
  }

  // 5. attention
  k_flash<<<(T / 16) * H_, 32, 0, stream>>>(query_b, key_b, v_b, attn_f, bptr, T);

  // 6. output projection
  cvt(attn_f, attn_bf, (size_t)T * DO);
  gemm(attn_bf, wo_bf, (float*)d_out, T, D, DO);
}
